// BasicIcoS2SUpBlock_76166950027366
// MI455X (gfx1250) — matrix operand order repaired and checked by reading, not compiled, not run on hardware
//
#include <hip/hip_runtime.h>
#include <hip/hip_bf16.h>

// ---------------------------------------------------------------------------
// BasicIcoS2SUpBlock on gfx1250: implicit-GEMM hex conv via v_wmma_f32_16x16x32_bf16
// Fully unrolled K/tap loops (templated Cin), bank-conflict-free padded LDS.
// ---------------------------------------------------------------------------

typedef __attribute__((ext_vector_type(16))) __bf16        v16bf;
typedef __attribute__((ext_vector_type(8)))  float         v8f;
typedef __attribute__((ext_vector_type(8)))  unsigned int  v8u;

#define B_    16
#define CIN_  64
#define COUT_ 32
#define H_    320
#define W_    128
#define HX_   160
#define WX_   64
#define HW_   (H_ * W_)
#define NB_STATS 64

__device__ __forceinline__ unsigned short f2bf(float f) {
  unsigned int u = __float_as_uint(f);
  unsigned int r = u + 0x7fffu + ((u >> 16) & 1u);   // round-to-nearest-even
  return (unsigned short)(r >> 16);
}

// --- weight prep: OIHW f32 -> [tap][co][Cin+2] bf16, hex mask folded in ----
template <int CIN>
__global__ void prep_w_kernel(const float* __restrict__ w,
                              unsigned short* __restrict__ wp) {
  constexpr int STR = CIN + 2;                        // padded row stride
  int n = 9 * COUT_ * STR;
  for (int i = blockIdx.x * blockDim.x + threadIdx.x; i < n;
       i += gridDim.x * blockDim.x) {
    int ci  = i % STR;
    int t   = i / STR;
    int co  = t % COUT_;
    int tap = t / COUT_;
    int ty = tap / 3, tx = tap % 3;
    float v = 0.0f;
    if (ci < CIN && !((ty == 0 && tx == 2) || (ty == 2 && tx == 0)))
      v = w[((co * CIN + ci) * 3 + ty) * 3 + tx];
    wp[i] = f2bf(v);
  }
}

// --- implicit-GEMM hex conv -------------------------------------------------
// MODE 0: input = x [B,CIN,160,64], read through nearest-neighbor 2x upsample
// MODE 1: input = f [B,32,320,128], BN(mean,rstd,gamma,beta)+ReLU fused on load
// Output tile per block: 32 channels x 64 pixels of one output row.
template <int CIN, int MODE>
__global__ __launch_bounds__(256)
void hexconv_wmma_kernel(const float* __restrict__ in,
                         const unsigned short* __restrict__ wp,
                         float* __restrict__ out,
                         const float* __restrict__ gamma,
                         const float* __restrict__ beta,
                         const float* __restrict__ mean,
                         const float* __restrict__ rstd) {
  constexpr int STR  = CIN + 2;                       // padded stride (33 dwords)
  constexpr int NKB  = CIN / 32;                      // K-blocks per tap
  constexpr int WELE = 9 * COUT_ * STR;               // weight elems in LDS

  extern __shared__ char smem[];
  unsigned short* Aw = (unsigned short*)smem;         // [9][32][STR]
  unsigned short* Bs = Aw + WELE;                     // [3*66][STR]

  const int tid = threadIdx.x;
  const int wtiles = W_ / 64;                         // 2
  int bid = blockIdx.x;
  int w0 = (bid % wtiles) * 64;
  int h0 = (bid / wtiles) % H_;
  int b  = bid / (wtiles * H_);

  // stage masked bf16 weights (u32 copies; prepped layout already padded)
  {
    const unsigned int* src = (const unsigned int*)wp;
    unsigned int* dst = (unsigned int*)Aw;
    constexpr int n32 = WELE >> 1;
    for (int i = tid; i < n32; i += 256) dst[i] = src[i];
  }

  // stage padded input slab: rows h0-1..h0+1 (wrap), cols w0-1..w0+64 (zero)
  {
    constexpr int total = 198 * CIN;                  // 3*66 positions x CIN ch
    for (int e = tid; e < total; e += 256) {
      int pc = e % 198;
      int ci = e / 198;
      int r = pc / 66, c = pc % 66;
      int hh = h0 - 1 + r;
      hh += (hh < 0) ? H_ : 0;
      hh -= (hh >= H_) ? H_ : 0;
      int ww = w0 - 1 + c;
      float v = 0.0f;
      if (ww >= 0 && ww < W_) {
        if (MODE == 0) {
          v = in[((b * CIN + ci) * HX_ + (hh >> 1)) * WX_ + (ww >> 1)];
        } else {
          float t = in[((b * COUT_ + ci) * H_ + hh) * W_ + ww];
          t = (t - mean[ci]) * rstd[ci] * gamma[ci] + beta[ci];
          v = t > 0.0f ? t : 0.0f;
        }
      }
      Bs[pc * STR + ci] = f2bf(v);
    }
  }
  __syncthreads();

  // one 16x16 output tile per wave32
  const int wave = tid >> 5;
  const int lane = tid & 31;
  const int half = lane >> 4;
  const int lan  = lane & 15;
  const int co0   = (wave >> 2) * 16;                 // M tile
  const int nbase = (wave & 3) * 16;                  // N tile

  const unsigned int* Aw32 = (const unsigned int*)Aw;
  const unsigned int* Bs32 = (const unsigned int*)Bs;

  // lane-invariant-except-lane base offsets (bf16 units)
  const int arow0 = (co0 + lan) * STR;                // + tap*COUT_*STR
  const int brow0 = (nbase + lan) * STR;              // + (ty*66+tx)*STR

  v8f acc = {};
#pragma unroll
  for (int tap = 0; tap < 9; ++tap) {
    constexpr int taps_y[9] = {0, 0, 0, 1, 1, 1, 2, 2, 2};
    constexpr int taps_x[9] = {0, 1, 2, 0, 1, 2, 0, 1, 2};
    const int brow = brow0 + (taps_y[tap] * 66 + taps_x[tap]) * STR;
    const int arow = arow0 + tap * COUT_ * STR;
#pragma unroll
    for (int kb = 0; kb < NKB; ++kb) {
      const int k0 = kb * 32;
      v8u au, bu;
#pragma unroll
      for (int j = 0; j < 8; ++j) {
        // A (16x32 bf16): VGPR0-3 -> K 0..7 / 8..15 by half; VGPR4-7 -> +16
        int ka = k0 + ((j < 4) ? (8 * half + 2 * j) : (16 + 8 * half + 2 * (j - 4)));
        au[j] = Aw32[(arow + ka) >> 1];
        // B (32x16 bf16): same K order as A: VGPR0-3 -> K 0..7 / 8..15 by half; VGPR4-7 -> +16
        int kbk = k0 + ((j < 4) ? (8 * half + 2 * j) : (16 + 8 * half + 2 * (j - 4)));
        bu[j] = Bs32[(brow + kbk) >> 1];
      }
      v16bf av = __builtin_bit_cast(v16bf, au);
      v16bf bv = __builtin_bit_cast(v16bf, bu);
      acc = __builtin_amdgcn_wmma_f32_16x16x32_bf16(
          false, av, false, bv, (short)0, acc, false, false);
    }
  }

  // C/D layout: VGPR r -> M = r + 8*half, N = lane&15
#pragma unroll
  for (int r = 0; r < 8; ++r) {
    int co = co0 + r + 8 * half;
    out[((b * COUT_ + co) * H_ + h0) * W_ + (w0 + nbase + lan)] = acc[r];
  }
}

// --- deterministic two-stage per-channel BN stats --------------------------
__global__ __launch_bounds__(256)
void stats_partial_kernel(const float* __restrict__ f,
                          float* __restrict__ psum, float* __restrict__ psq) {
  int ch   = blockIdx.x / NB_STATS;
  int part = blockIdx.x % NB_STATS;
  const int P = B_ * HW_;
  const int chunk = P / NB_STATS;
  int start = part * chunk;
  float s = 0.0f, q = 0.0f;
  for (int p = start + threadIdx.x; p < start + chunk; p += 256) {
    int bb = p / HW_;
    int rem = p - bb * HW_;
    float v = f[(bb * COUT_ + ch) * HW_ + rem];
    s += v;
    q += v * v;
  }
  __shared__ float ss[256];
  __shared__ float sq[256];
  ss[threadIdx.x] = s; sq[threadIdx.x] = q;
  __syncthreads();
  for (int st = 128; st > 0; st >>= 1) {
    if (threadIdx.x < st) {
      ss[threadIdx.x] += ss[threadIdx.x + st];
      sq[threadIdx.x] += sq[threadIdx.x + st];
    }
    __syncthreads();
  }
  if (threadIdx.x == 0) {
    psum[ch * NB_STATS + part] = ss[0];
    psq [ch * NB_STATS + part] = sq[0];
  }
}

__global__ void stats_final_kernel(const float* __restrict__ psum,
                                   const float* __restrict__ psq,
                                   float* __restrict__ mean,
                                   float* __restrict__ rstd) {
  int ch = threadIdx.x;                               // 32 threads
  if (ch >= COUT_) return;
  float s = 0.0f, q = 0.0f;
  for (int i = 0; i < NB_STATS; ++i) {
    s += psum[ch * NB_STATS + i];
    q += psq [ch * NB_STATS + i];
  }
  const float Pn = (float)(B_ * HW_);
  float m = s / Pn;
  float var = q / Pn - m * m;
  mean[ch] = m;
  rstd[ch] = rsqrtf(var + 1e-5f);
}

// --- final: relu(BN(out0raw) + BN(out1raw)) --------------------------------
__global__ __launch_bounds__(256)
void combine_kernel(const float* __restrict__ a, const float* __restrict__ bsrc,
                    const float* __restrict__ g01, const float* __restrict__ b01,
                    const float* __restrict__ m0,  const float* __restrict__ r0,
                    const float* __restrict__ g10, const float* __restrict__ b10,
                    const float* __restrict__ m1,  const float* __restrict__ r1,
                    float* __restrict__ out) {
  const long long n = (long long)B_ * COUT_ * HW_;
  for (long long i = blockIdx.x * 256ll + threadIdx.x; i < n;
       i += (long long)gridDim.x * 256ll) {
    int ch = (int)((i / HW_) % COUT_);
    float v0 = (a[i]    - m0[ch]) * r0[ch] * g01[ch] + b01[ch];
    float v1 = (bsrc[i] - m1[ch]) * r1[ch] * g10[ch] + b10[ch];
    float v = v0 + v1;
    out[i] = v > 0.0f ? v : 0.0f;
  }
}

// ---------------------------------------------------------------------------
extern "C" void kernel_launch(void* const* d_in, const int* in_sizes, int n_in,
                              void* d_out, int out_size, void* d_ws, size_t ws_size,
                              hipStream_t stream) {
  const float* x   = (const float*)d_in[0];
  const float* w00 = (const float*)d_in[1];
  const float* w01 = (const float*)d_in[2];
  const float* w10 = (const float*)d_in[3];
  const float* g00 = (const float*)d_in[4];
  const float* b00 = (const float*)d_in[5];
  const float* g01 = (const float*)d_in[6];
  const float* b01 = (const float*)d_in[7];
  const float* g10 = (const float*)d_in[8];
  const float* b10 = (const float*)d_in[9];
  float* out = (float*)d_out;

  const size_t N1 = (size_t)B_ * COUT_ * HW_;         // 20,971,520 floats
  float* wsA = (float*)d_ws;                          // conv1raw, later out1raw
  float* wsB = wsA + N1;                              // out0raw
  char* p = (char*)(wsB + N1);
  unsigned short* wp0 = (unsigned short*)p; p += 9 * COUT_ * (CIN_  + 2) * 2;
  unsigned short* wp1 = (unsigned short*)p; p += 9 * COUT_ * (COUT_ + 2) * 2;
  unsigned short* wp2 = (unsigned short*)p; p += 9 * COUT_ * (CIN_  + 2) * 2;
  float* psum   = (float*)p; p += COUT_ * NB_STATS * 4;
  float* psq    = (float*)p; p += COUT_ * NB_STATS * 4;
  float* mean00 = (float*)p; p += COUT_ * 4;
  float* rstd00 = (float*)p; p += COUT_ * 4;
  float* mean01 = (float*)p; p += COUT_ * 4;
  float* rstd01 = (float*)p; p += COUT_ * 4;
  float* mean10 = (float*)p; p += COUT_ * 4;
  float* rstd10 = (float*)p; p += COUT_ * 4;

  const int convBlocks = B_ * H_ * (W_ / 64);         // 10240
  const size_t shmem1 = (size_t)(9 * COUT_ + 198) * (CIN_  + 2) * 2;  // 64,152 B
  const size_t shmem2 = (size_t)(9 * COUT_ + 198) * (COUT_ + 2) * 2;  // 33,048 B

  // weight prep (hex mask folded in, bf16, padded [tap][co][Cin+2])
  prep_w_kernel<CIN_ ><<<19, 256, 0, stream>>>(w00, wp0);
  prep_w_kernel<COUT_><<<10, 256, 0, stream>>>(w01, wp1);
  prep_w_kernel<CIN_ ><<<19, 256, 0, stream>>>(w10, wp2);

  // conv1: up(x) * w00 -> wsA ; stats
  hexconv_wmma_kernel<CIN_, 0><<<convBlocks, 256, shmem1, stream>>>(
      x, wp0, wsA, nullptr, nullptr, nullptr, nullptr);
  stats_partial_kernel<<<COUT_ * NB_STATS, 256, 0, stream>>>(wsA, psum, psq);
  stats_final_kernel<<<1, 32, 0, stream>>>(psum, psq, mean00, rstd00);

  // conv2: relu(BN(wsA)) * w01 -> wsB ; stats
  hexconv_wmma_kernel<COUT_, 1><<<convBlocks, 256, shmem2, stream>>>(
      wsA, wp1, wsB, g00, b00, mean00, rstd00);
  stats_partial_kernel<<<COUT_ * NB_STATS, 256, 0, stream>>>(wsB, psum, psq);
  stats_final_kernel<<<1, 32, 0, stream>>>(psum, psq, mean01, rstd01);

  // conv3: up(x) * w10 -> wsA (reuse) ; stats
  hexconv_wmma_kernel<CIN_, 0><<<convBlocks, 256, shmem1, stream>>>(
      x, wp2, wsA, nullptr, nullptr, nullptr, nullptr);
  stats_partial_kernel<<<COUT_ * NB_STATS, 256, 0, stream>>>(wsA, psum, psq);
  stats_final_kernel<<<1, 32, 0, stream>>>(psum, psq, mean10, rstd10);

  // out = relu(BN(wsB) + BN(wsA))
  combine_kernel<<<20480, 256, 0, stream>>>(
      wsB, wsA, g01, b01, mean01, rstd01, g10, b10, mean10, rstd10, out);
}
